// GLCN_70489003262649
// MI455X (gfx1250) — compile-verified, matrix operand order checked by reading, not run on hardware
//
#include <hip/hip_runtime.h>

// ---------------- problem constants (match reference) ----------------
#define N_NODES 8192
#define F_DIM   256
#define D_DIM   128
#define SLOPE   0.2f

typedef __attribute__((ext_vector_type(16))) __bf16       v16bf;
typedef __attribute__((ext_vector_type(8)))  float        v8f;
typedef __attribute__((ext_vector_type(4)))  unsigned int u32x4;
typedef unsigned short u16;

// ---------------- helpers ----------------
__device__ __forceinline__ u16 f2bf_rne(float x) {
    unsigned u = __float_as_uint(x);
    unsigned r = 0x7FFFu + ((u >> 16) & 1u);
    return (u16)((u + r) >> 16);
}
__device__ __forceinline__ float bf2f(u16 h) {
    return __uint_as_float(((unsigned)h) << 16);
}
// order-preserving float -> uint encoding (for atomicMax)
__device__ __forceinline__ unsigned enc_f32(float f) {
    unsigned u = __float_as_uint(f);
    return (u & 0x80000000u) ? ~u : (u | 0x80000000u);
}
__device__ __forceinline__ float dec_f32(unsigned e) {
    unsigned u = (e & 0x80000000u) ? (e ^ 0x80000000u) : ~e;
    return __uint_as_float(u);
}
__device__ __forceinline__ float leaky(float x) { return x > 0.f ? x : SLOPE * x; }

// ---------------- kernels ----------------

// grid-stride fill
__global__ void k_fill_u32(unsigned* __restrict__ p, unsigned v, int n) {
    for (int i = blockIdx.x * blockDim.x + threadIdx.x; i < n; i += gridDim.x * blockDim.x)
        p[i] = v;
}

// split fp32 -> bf16 hi/lo (row-major, used for X)
__global__ void k_split(const float* __restrict__ src, u16* __restrict__ hi,
                        u16* __restrict__ lo, int n) {
    int i = blockIdx.x * blockDim.x + threadIdx.x;
    if (i >= n) return;
    float x = src[i];
    u16 h = f2bf_rne(x);
    hi[i] = h;
    lo[i] = f2bf_rne(x - bf2f(h));
}

// split + transpose Ws [F x D] -> WT [D x F] bf16 hi/lo (so B columns load contiguously)
__global__ void k_split_T(const float* __restrict__ src, u16* __restrict__ hiT,
                          u16* __restrict__ loT) {
    int i = blockIdx.x * blockDim.x + threadIdx.x;
    if (i >= F_DIM * D_DIM) return;
    int k = i / D_DIM, n = i % D_DIM;
    float x = src[i];
    u16 h = f2bf_rne(x);
    hiT[n * F_DIM + k] = h;
    loT[n * F_DIM + k] = f2bf_rne(x - bf2f(h));
}

// Wh = X @ Ws via v_wmma_f32_16x16x32_bf16 with bf16x3 compensation.
// One wave -> one 16x16 tile of C; 4 waves/block over M; grid = (512/4, 128/16).
__global__ void __launch_bounds__(128)
k_wmma_gemm(const u16* __restrict__ Ahi, const u16* __restrict__ Alo,
            const u16* __restrict__ BThi, const u16* __restrict__ BTlo,
            float* __restrict__ C) {
    const int lane   = threadIdx.x & 31;
    const int wave   = threadIdx.x >> 5;
    const int tile_m = blockIdx.x * 4 + wave;   // 0..511
    const int tile_n = blockIdx.y;              // 0..7
    const int half   = lane >> 4;               // K-half selector per ISA layout
    const int idx16  = lane & 15;               // row (A) / col (B) within tile

    const u16* arh = Ahi  + (size_t)(tile_m * 16 + idx16) * F_DIM;
    const u16* arl = Alo  + (size_t)(tile_m * 16 + idx16) * F_DIM;
    const u16* brh = BThi + (size_t)(tile_n * 16 + idx16) * F_DIM;
    const u16* brl = BTlo + (size_t)(tile_n * 16 + idx16) * F_DIM;

    v8f acc = {};
#pragma unroll
    for (int k0 = 0; k0 < F_DIM; k0 += 32) {
        const int o0 = k0 + half * 8;        // K = k0 + {0..7 | 8..15}
        const int o1 = k0 + 16 + half * 8;   // K = k0 + {16..23 | 24..31}
        v16bf ah, al, bh, bl;
        {
            u32x4* p = (u32x4*)&ah;
            p[0] = *(const u32x4*)(arh + o0); p[1] = *(const u32x4*)(arh + o1);
        }
        {
            u32x4* p = (u32x4*)&al;
            p[0] = *(const u32x4*)(arl + o0); p[1] = *(const u32x4*)(arl + o1);
        }
        {
            u32x4* p = (u32x4*)&bh;
            p[0] = *(const u32x4*)(brh + o0); p[1] = *(const u32x4*)(brh + o1);
        }
        {
            u32x4* p = (u32x4*)&bl;
            p[0] = *(const u32x4*)(brl + o0); p[1] = *(const u32x4*)(brl + o1);
        }
        acc = __builtin_amdgcn_wmma_f32_16x16x32_bf16(false, ah, false, bh, (short)0, acc, false, false);
        acc = __builtin_amdgcn_wmma_f32_16x16x32_bf16(false, ah, false, bl, (short)0, acc, false, false);
        acc = __builtin_amdgcn_wmma_f32_16x16x32_bf16(false, al, false, bh, (short)0, acc, false, false);
    }
    // C/D layout: VGPR r, lanes0-15 -> M=r, lanes16-31 -> M=r+8; N = lane&15
    float* crow = C + (size_t)(tile_m * 16 + half * 8) * D_DIM + tile_n * 16 + idx16;
#pragma unroll
    for (int r = 0; r < 8; r++) crow[(size_t)r * D_DIM] = acc[r];
}

// s1[i] = Wh[i,:].a[0:D], s2[i] = Wh[i,:].a[D:2D]; one wave per row
__global__ void k_scores(const float* __restrict__ Wh, const float* __restrict__ avec,
                         float* __restrict__ s1, float* __restrict__ s2) {
    int row  = blockIdx.x * (blockDim.x >> 5) + (threadIdx.x >> 5);
    int lane = threadIdx.x & 31;
    const float* wr = Wh + (size_t)row * D_DIM;
    float v1 = 0.f, v2 = 0.f;
#pragma unroll
    for (int c = 0; c < 4; c++) {
        int col = lane + 32 * c;
        float w = wr[col];
        v1 += w * avec[col];
        v2 += w * avec[D_DIM + col];
    }
#pragma unroll
    for (int off = 16; off > 0; off >>= 1) {
        v1 += __shfl_xor(v1, off);
        v2 += __shfl_xor(v2, off);
    }
    if (lane == 0) { s1[row] = v1; s2[row] = v2; }
}

// pass A: dedupe edges via adjacency bitmask (first writer owns) + row max of scores
__global__ void k_edge_max(const int* __restrict__ es, const int* __restrict__ et,
                           const float* __restrict__ s1, const float* __restrict__ s2,
                           unsigned* __restrict__ bitmask, unsigned char* __restrict__ owner,
                           unsigned* __restrict__ rowmax, int E) {
    int e = blockIdx.x * blockDim.x + threadIdx.x;
    if (e >= E) return;
    int s = es[e], t = et[e];
    unsigned pos = (unsigned)s * N_NODES + (unsigned)t;   // < 2^26
    unsigned bit = 1u << (pos & 31u);
    unsigned old = atomicOr(&bitmask[pos >> 5], bit);
    bool own = !(old & bit);
    owner[e] = (unsigned char)own;
    if (own) {
        float ev = leaky(s1[s] + s2[t]);
        atomicMax(&rowmax[s], enc_f32(ev));
    }
}

// pass B: softmax denominator per row (owner edges only)
__global__ void k_edge_denom(const int* __restrict__ es, const int* __restrict__ et,
                             const unsigned char* __restrict__ owner,
                             const float* __restrict__ s1, const float* __restrict__ s2,
                             const unsigned* __restrict__ rowmax,
                             float* __restrict__ denom, int E) {
    int e = blockIdx.x * blockDim.x + threadIdx.x;
    if (e >= E || !owner[e]) return;
    int s = es[e], t = et[e];
    float ev = leaky(s1[s] + s2[t]);
    atomicAdd(&denom[s], __expf(ev - dec_f32(rowmax[s])));
}

// pass C: out[s,:] += att * Wh[t,:]; one wave per edge, 4 cols/lane
__global__ void k_edge_accum(const int* __restrict__ es, const int* __restrict__ et,
                             const unsigned char* __restrict__ owner,
                             const float* __restrict__ s1, const float* __restrict__ s2,
                             const unsigned* __restrict__ rowmax, const float* __restrict__ denom,
                             const float* __restrict__ Wh, float* __restrict__ out, int E) {
    int e    = (blockIdx.x * blockDim.x + threadIdx.x) >> 5;  // wave-uniform
    int lane = threadIdx.x & 31;
    if (e >= E || !owner[e]) return;
    int s = es[e], t = et[e];
    const float* wr = Wh + (size_t)t * D_DIM;
    __builtin_prefetch(wr, 0, 0);
    float ev = leaky(s1[s] + s2[t]);
    float w  = __expf(ev - dec_f32(rowmax[s])) / denom[s];
    float* orow = out + (size_t)s * D_DIM;
#pragma unroll
    for (int c = 0; c < 4; c++) {
        int col = lane + 32 * c;
        atomicAdd(&orow[col], w * wr[col]);
    }
}

// final elementwise ELU
__global__ void k_elu(float* __restrict__ out, int n) {
    int i = blockIdx.x * blockDim.x + threadIdx.x;
    if (i < n) {
        float x = out[i];
        out[i] = x > 0.f ? x : __expf(x) - 1.f;
    }
}

// ---------------- launch ----------------
extern "C" void kernel_launch(void* const* d_in, const int* in_sizes, int n_in,
                              void* d_out, int out_size, void* d_ws, size_t ws_size,
                              hipStream_t stream) {
    const int*   edge = (const int*)d_in[0];   // [2, E] flat: sources then targets
    const float* X    = (const float*)d_in[1]; // [N, F]
    const float* Ws   = (const float*)d_in[2]; // [F, D]
    const float* avec = (const float*)d_in[3]; // [2D, 1]
    float* out = (float*)d_out;                // [N, D]

    const int E = in_sizes[0] / 2;
    const int* es = edge;
    const int* et = edge + E;

    // ---- scratch carve-up (256B aligned regions) ----
    char* base = (char*)d_ws;
    size_t off = 0;
    auto carve = [&](size_t bytes) -> void* {
        void* p = base + off;
        off += (bytes + 255) & ~(size_t)255;
        return p;
    };
    float*    Wh      = (float*)   carve((size_t)N_NODES * D_DIM * 4);
    u16*      Xhi     = (u16*)     carve((size_t)N_NODES * F_DIM * 2);
    u16*      Xlo     = (u16*)     carve((size_t)N_NODES * F_DIM * 2);
    u16*      WThi    = (u16*)     carve((size_t)D_DIM * F_DIM * 2);
    u16*      WTlo    = (u16*)     carve((size_t)D_DIM * F_DIM * 2);
    float*    s1      = (float*)   carve((size_t)N_NODES * 4);
    float*    s2      = (float*)   carve((size_t)N_NODES * 4);
    unsigned* rowmax  = (unsigned*)carve((size_t)N_NODES * 4);
    float*    denom   = (float*)   carve((size_t)N_NODES * 4);
    unsigned* bitmask = (unsigned*)carve(((size_t)N_NODES * N_NODES / 32) * 4);
    unsigned char* owner = (unsigned char*)carve((size_t)E);

    // ---- re-init all accumulation state every call (harness doesn't re-poison) ----
    {
        float ninf = -1e30f;
        unsigned u; __builtin_memcpy(&u, &ninf, 4);
        unsigned enc_ninf = ~u;                     // encoded -1e30 (negative branch)
        int nbm = N_NODES * N_NODES / 32;
        k_fill_u32<<<4096, 256, 0, stream>>>(bitmask, 0u, nbm);
        k_fill_u32<<<(N_NODES + 255) / 256, 256, 0, stream>>>(rowmax, enc_ninf, N_NODES);
        k_fill_u32<<<(N_NODES + 255) / 256, 256, 0, stream>>>((unsigned*)denom, 0u, N_NODES);
        k_fill_u32<<<(N_NODES * D_DIM + 255) / 256, 256, 0, stream>>>((unsigned*)out, 0u, N_NODES * D_DIM);
    }

    // ---- bf16 hi/lo splits ----
    k_split  <<<(N_NODES * F_DIM + 255) / 256, 256, 0, stream>>>(X, Xhi, Xlo, N_NODES * F_DIM);
    k_split_T<<<(F_DIM * D_DIM + 255) / 256, 256, 0, stream>>>(Ws, WThi, WTlo);

    // ---- Wh = X @ Ws (WMMA bf16x3) ----
    {
        dim3 grid(N_NODES / 16 / 4, D_DIM / 16);   // (128, 8)
        k_wmma_gemm<<<grid, 128, 0, stream>>>(Xhi, Xlo, WThi, WTlo, Wh);
    }

    // ---- per-row score components ----
    k_scores<<<N_NODES / 8, 256, 0, stream>>>(Wh, avec, s1, s2);

    // ---- sparse masked softmax + aggregation over edges ----
    k_edge_max  <<<(E + 255) / 256, 256, 0, stream>>>(es, et, s1, s2, bitmask, owner, rowmax, E);
    k_edge_denom<<<(E + 255) / 256, 256, 0, stream>>>(es, et, owner, s1, s2, rowmax, denom, E);
    k_edge_accum<<<(E * 32 + 255) / 256, 256, 0, stream>>>(es, et, owner, s1, s2, rowmax, denom, Wh, out, E);

    // ---- ELU ----
    k_elu<<<(N_NODES * D_DIM + 255) / 256, 256, 0, stream>>>(out, N_NODES * D_DIM);

    (void)n_in; (void)out_size; (void)ws_size;
}
